// CapsuleLayer_6743098655303
// MI455X (gfx1250) — matrix operand order repaired and checked by reading, not compiled, not run on hardware
//
#include <hip/hip_runtime.h>
#include <hip/hip_bf16.h>
#include <math.h>

// ---------------------------------------------------------------------------
// CapsuleLayer on MI455X (gfx1250): bf16x3-split WMMA GEMMs + TDM async tile
// staging + L2-resident dynamic routing.
// ---------------------------------------------------------------------------

typedef __attribute__((ext_vector_type(8)))  float        v8f;
typedef __attribute__((ext_vector_type(16))) __bf16       v16bf;
typedef __attribute__((ext_vector_type(4)))  unsigned int u32x4;
typedef __attribute__((ext_vector_type(8)))  int          i32x8;
typedef __attribute__((ext_vector_type(4)))  int          i32x4;

#define BATCH   256
#define NCAP    128
#define DIN     768
#define DCAP    32
#define NOUT    4096          // NCAP * DCAP
#define MROWS   32768         // BATCH * NCAP
#define KATT    98304         // NCAP * DIN

// D = A*B + C with f32 accumulation; A reconstructed as (ah+al), B as (bh+bl),
// dropping the al*bl term (3xBF16 ~ fp32-class precision).
__device__ __forceinline__ v8f wmma_bf16x3(v16bf ah, v16bf al, v16bf bh,
                                           v16bf bl, v8f acc) {
  acc = __builtin_amdgcn_wmma_f32_16x16x32_bf16(false, ah, false, bh,
                                                (short)0, acc, false, false);
  acc = __builtin_amdgcn_wmma_f32_16x16x32_bf16(false, ah, false, bl,
                                                (short)0, acc, false, false);
  acc = __builtin_amdgcn_wmma_f32_16x16x32_bf16(false, al, false, bh,
                                                (short)0, acc, false, false);
  return acc;
}

// ---------------------------------------------------------------------------
// Tensor Data Mover: 2D tile (tile_k x tile_rows) of f32 from a row-major
// (tensor_rows x tensor_k, stride row_stride) tensor into LDS, with optional
// LDS row padding (ISA 8.3/8.4 D# layout, tracked by TENSORcnt).
// 6-arg builtin form (clang-23 / therock-10.0 headers).
// ---------------------------------------------------------------------------
__device__ __forceinline__ void tdm_load_2d(unsigned lds_off, const void* gptr,
                                            unsigned tile_k, unsigned tile_rows,
                                            unsigned tensor_k,
                                            unsigned tensor_rows,
                                            unsigned row_stride,
                                            unsigned pad_interval_code,
                                            unsigned pad_amount_code,
                                            unsigned pad_enable) {
  unsigned long long ga = (unsigned long long)gptr;
  u32x4 g0;
  g0[0] = 1u;                                   // count=1, user descriptor
  g0[1] = lds_off;                              // lds_addr [63:32]
  g0[2] = (unsigned)ga;                         // global_addr [95:64]
  g0[3] = (unsigned)((ga >> 32) & 0x01FFFFFFull) | 0x80000000u;  // type=2
  i32x8 g1;
  g1[0] = (int)((2u << 16) |                    // data_size = 4B
                (pad_enable << 20) | (pad_interval_code << 22) |
                (pad_amount_code << 25));
  g1[1] = (int)((tensor_k & 0xFFFFu) << 16);                       // dim0 lo16
  g1[2] = (int)(((tensor_k >> 16) & 0xFFFFu) |
                ((tensor_rows & 0xFFFFu) << 16));                  // dim0 hi / dim1 lo
  g1[3] = (int)(((tensor_rows >> 16) & 0xFFFFu) |
                ((tile_k & 0xFFFFu) << 16));                       // dim1 hi / tile_dim0
  g1[4] = (int)(tile_rows & 0xFFFFu);                              // tile_dim1
  g1[5] = (int)row_stride;                                         // dim0_stride lo32
  g1[6] = 0;
  g1[7] = 0;
  i32x4 z4;
  z4[0] = z4[1] = z4[2] = z4[3] = 0;             // 2D tensor: groups 2/3 unused
  i32x8 z8;
  z8[0] = z8[1] = z8[2] = z8[3] = 0;
  z8[4] = z8[5] = z8[6] = z8[7] = 0;
  __builtin_amdgcn_tensor_load_to_lds(g0, g1, z4, z4, z8, 0);
}

// ---------------------------------------------------------------------------
// Kernel 1: r[256,128] = relu(flat(x) @ l1_w^T + l1_b).
// One block per 16x16 output tile; 8 waves split K=98304, bf16x3 WMMA,
// cross-wave LDS reduction.
// ---------------------------------------------------------------------------
__global__ __launch_bounds__(256) void attn1_kernel(
    const float* __restrict__ xf, const float* __restrict__ l1w,
    const float* __restrict__ l1b, float* __restrict__ r) {
  __shared__ float part[8][32][8];
  const int tid = threadIdx.x;
  const int wid = tid >> 5, lane = tid & 31;
  const int lrow = lane & 15, khalf = lane >> 4;
  const int mb = blockIdx.y, nb = blockIdx.x;

  const float* arow = xf + (size_t)(mb * 16 + lrow) * KATT;
  const float* brow = l1w + (size_t)(nb * 16 + lrow) * KATT;

  v8f acc = {0.f, 0.f, 0.f, 0.f, 0.f, 0.f, 0.f, 0.f};
  const int kbeg = wid * (KATT / 8);
  const int kend = kbeg + (KATT / 8);
  for (int k0 = kbeg; k0 < kend; k0 += 32) {
    v16bf ah, al, bh, bl;
    // A fragment 16x32: lanes0-15 K=0..15, lanes16-31 K=16..31 (matches B)
#pragma unroll
    for (int jp = 0; jp < 8; ++jp) {
      const int k = khalf * 16 + jp * 2;
      const float2 w = *(const float2*)(arow + k0 + k);
      const __bf16 hx = (__bf16)w.x;
      ah[2 * jp] = hx;
      al[2 * jp] = (__bf16)(w.x - (float)hx);
      const __bf16 hy = (__bf16)w.y;
      ah[2 * jp + 1] = hy;
      al[2 * jp + 1] = (__bf16)(w.y - (float)hy);
    }
    // B fragment 32x16: lanes0-15 K=0..15, lanes16-31 K=16..31, N=lane&15
#pragma unroll
    for (int q = 0; q < 4; ++q) {
      const float4 w = *(const float4*)(brow + k0 + khalf * 16 + 4 * q);
      const __bf16 h0 = (__bf16)w.x;
      bh[4 * q + 0] = h0; bl[4 * q + 0] = (__bf16)(w.x - (float)h0);
      const __bf16 h1 = (__bf16)w.y;
      bh[4 * q + 1] = h1; bl[4 * q + 1] = (__bf16)(w.y - (float)h1);
      const __bf16 h2 = (__bf16)w.z;
      bh[4 * q + 2] = h2; bl[4 * q + 2] = (__bf16)(w.z - (float)h2);
      const __bf16 h3 = (__bf16)w.w;
      bh[4 * q + 3] = h3; bl[4 * q + 3] = (__bf16)(w.w - (float)h3);
    }
    acc = wmma_bf16x3(ah, al, bh, bl, acc);
  }
#pragma unroll
  for (int rr = 0; rr < 8; ++rr) part[wid][lane][rr] = acc[rr];
  __syncthreads();
  // Reduce over waves; C layout: M = r + 8*(lane>=16), N = lane&15
  const int lane2 = tid & 31, rr2 = tid >> 5;
  float ssum = 0.f;
#pragma unroll
  for (int w = 0; w < 8; ++w) ssum += part[w][lane2][rr2];
  const int M = rr2 + ((lane2 >> 4) << 3);
  const int N = lane2 & 15;
  const int gm = mb * 16 + M, gn = nb * 16 + N;
  ssum += l1b[gn];
  r[gm * NCAP + gn] = fmaxf(ssum, 0.f);
}

// ---------------------------------------------------------------------------
// Kernel 2: g[b,c] = 1 + tanh(r[b,:] @ l2_w[c,:] + l2_b[c])  (tiny, VALU)
// ---------------------------------------------------------------------------
__global__ __launch_bounds__(128) void attn2_kernel(
    const float* __restrict__ r, const float* __restrict__ l2w,
    const float* __restrict__ l2b, float* __restrict__ g) {
  __shared__ float rs[NCAP];
  const int b = blockIdx.x, c = threadIdx.x;
  rs[c] = r[b * NCAP + c];
  __syncthreads();
  float acc = l2b[c];
  const float* wrow = l2w + c * NCAP;
  for (int k = 0; k < NCAP; ++k) acc = fmaf(rs[k], wrow[k], acc);
  g[b * NCAP + c] = 1.f + tanhf(acc);
}

// ---------------------------------------------------------------------------
// Kernel 3: p[m, o] = g[m] * (x[m,:] @ W[o,:]) + bias[m%128, o]
// Block tile 64x128, 8 waves (2x4), each wave 32x32 = 2x2 WMMA tiles.
// Double-buffered TDM tile staging (A 64x32, B 128x32 f32), LDS-padded.
// ---------------------------------------------------------------------------
#define ASTRIDE 34   // 32 dwords + 2 pad (TDM pad_interval=32, amount=2)
#define BSTRIDE 36   // 32 dwords + 4 pad

__global__ __launch_bounds__(256) void p_gemm_kernel(
    const float* __restrict__ x, const float* __restrict__ W,
    const float* __restrict__ bias, const float* __restrict__ g,
    float* __restrict__ p) {
  __shared__ __align__(16) float As[2][64 * ASTRIDE];
  __shared__ __align__(16) float Bs[2][128 * BSTRIDE];
  const int tid = threadIdx.x;
  const int wid = tid >> 5, lane = tid & 31;
  const int wm = wid >> 2, wn = wid & 3;
  const int bn = blockIdx.x, bm = blockIdx.y;
  const int lrow = lane & 15, khalf = lane >> 4;

  const v8f vz = {0.f, 0.f, 0.f, 0.f, 0.f, 0.f, 0.f, 0.f};
  v8f acc[2][2];
  acc[0][0] = vz; acc[0][1] = vz; acc[1][0] = vz; acc[1][1] = vz;

  auto stage = [&](int buf, int k0) {
    if (wid == 0) {  // one wave issues the TDM descriptors (EXEC ignored)
      tdm_load_2d((unsigned)(size_t)&As[buf][0],
                  x + (size_t)bm * 64 * DIN + k0,
                  32, 64, DIN, MROWS, DIN, /*interval 32dw*/ 4, /*+2dw*/ 1, 1);
      tdm_load_2d((unsigned)(size_t)&Bs[buf][0],
                  W + (size_t)bn * 128 * DIN + k0,
                  32, 128, DIN, NOUT, DIN, /*interval 32dw*/ 4, /*+4dw*/ 3, 1);
    }
  };

  stage(0, 0);
  if (wid == 0) __builtin_amdgcn_s_wait_tensorcnt(0);
  __syncthreads();

  for (int t = 0; t < DIN / 32; ++t) {
    const int buf = t & 1;
    if (t + 1 < DIN / 32) stage(buf ^ 1, (t + 1) * 32);  // overlap with compute

    v16bf ah[2], al[2], bh[2], bl[2];
#pragma unroll
    for (int mt = 0; mt < 2; ++mt) {
      const float* src = &As[buf][(wm * 32 + mt * 16 + lrow) * ASTRIDE];
#pragma unroll
      for (int jp = 0; jp < 8; ++jp) {
        const int k = khalf * 16 + jp * 2;
        const float2 w = *(const float2*)(src + k);
        const __bf16 hx = (__bf16)w.x;
        ah[mt][2 * jp] = hx;
        al[mt][2 * jp] = (__bf16)(w.x - (float)hx);
        const __bf16 hy = (__bf16)w.y;
        ah[mt][2 * jp + 1] = hy;
        al[mt][2 * jp + 1] = (__bf16)(w.y - (float)hy);
      }
    }
#pragma unroll
    for (int nt = 0; nt < 2; ++nt) {
      const float* src =
          &Bs[buf][(wn * 32 + nt * 16 + lrow) * BSTRIDE + khalf * 16];
#pragma unroll
      for (int q = 0; q < 4; ++q) {
        const float4 w = *(const float4*)(src + 4 * q);
        const __bf16 h0 = (__bf16)w.x;
        bh[nt][4 * q + 0] = h0; bl[nt][4 * q + 0] = (__bf16)(w.x - (float)h0);
        const __bf16 h1 = (__bf16)w.y;
        bh[nt][4 * q + 1] = h1; bl[nt][4 * q + 1] = (__bf16)(w.y - (float)h1);
        const __bf16 h2 = (__bf16)w.z;
        bh[nt][4 * q + 2] = h2; bl[nt][4 * q + 2] = (__bf16)(w.z - (float)h2);
        const __bf16 h3 = (__bf16)w.w;
        bh[nt][4 * q + 3] = h3; bl[nt][4 * q + 3] = (__bf16)(w.w - (float)h3);
      }
    }
#pragma unroll
    for (int mt = 0; mt < 2; ++mt)
#pragma unroll
      for (int nt = 0; nt < 2; ++nt)
        acc[mt][nt] = wmma_bf16x3(ah[mt], al[mt], bh[nt], bl[nt], acc[mt][nt]);

    if (wid == 0) __builtin_amdgcn_s_wait_tensorcnt(0);
    __syncthreads();  // next buffer ready; everyone done reading current
  }

  // Epilogue: row gate + bias, streaming (non-temporal) stores of p (512 MB).
#pragma unroll
  for (int mt = 0; mt < 2; ++mt) {
#pragma unroll
    for (int nt = 0; nt < 2; ++nt) {
      const int rowbase = bm * 64 + wm * 32 + mt * 16 + khalf * 8;
      const int col = bn * 128 + wn * 32 + nt * 16 + lrow;
#pragma unroll
      for (int rr = 0; rr < 8; ++rr) {
        const int rm = rowbase + rr;
        const float val =
            acc[mt][nt][rr] * g[rm] + bias[(rm & (NCAP - 1)) * NOUT + col];
        __builtin_nontemporal_store(val, p + (size_t)rm * NOUT + col);
      }
    }
  }
}

// ---------------------------------------------------------------------------
// Kernel 4: dynamic routing, one block per batch element (2 MB p-slice stays
// L2-resident across the 5 passes). out = clamp(nn/(0.5+nn), ...) where
// nn = ||s||^2 (norm of squash(s) simplifies).
// ---------------------------------------------------------------------------
__global__ __launch_bounds__(256) void routing_kernel(
    const float* __restrict__ p, float* __restrict__ out) {
  extern __shared__ float sm[];
  float* Blog = sm;               // 128*128 routing logits
  float* Beta = sm + 16384;       // 128*128 softmax
  float* CC   = sm + 32768;       // 128*32 squashed capsules
  float* Red  = sm + 36864;       // 256 reduction scratch
  const int t = threadIdx.x;
  const float* pb = p + (size_t)blockIdx.x * (NCAP * NCAP * DCAP);

  for (int i = t; i < NCAP * NCAP; i += 256) Blog[i] = 0.f;
  __syncthreads();

  const int j = t >> 1;            // output capsule handled by this thread
  const int db = (t & 1) * 16;     // half of DCAP

  for (int it = 0; it < 3; ++it) {
    // softmax over j for each input capsule row n (threads 0..127)
    if (t < NCAP) {
      const float* row = Blog + t * NCAP;
      float m = row[0];
      for (int q = 1; q < NCAP; ++q) m = fmaxf(m, row[q]);
      float sum = 0.f;
      for (int q = 0; q < NCAP; ++q) {
        const float e = expf(row[q] - m);
        Beta[t * NCAP + q] = e;
        sum += e;
      }
      const float inv = 1.f / sum;
      for (int q = 0; q < NCAP; ++q) Beta[t * NCAP + q] *= inv;
    }
    __syncthreads();

    // s[j, db..db+16) = sum_n beta[n,j] * p[n,j,:]
    float s[16];
#pragma unroll
    for (int i = 0; i < 16; ++i) s[i] = 0.f;
    for (int n = 0; n < NCAP; ++n) {
      const float w = Beta[n * NCAP + j];
      const float4* pr =
          (const float4*)(pb + ((size_t)(n * NCAP + j)) * DCAP + db);
#pragma unroll
      for (int q = 0; q < 4; ++q) {
        const float4 v = pr[q];
        s[4 * q + 0] += w * v.x;
        s[4 * q + 1] += w * v.y;
        s[4 * q + 2] += w * v.z;
        s[4 * q + 3] += w * v.w;
      }
    }
    float ss = 0.f;
#pragma unroll
    for (int i = 0; i < 16; ++i) ss += s[i] * s[i];
    Red[t] = ss;
    __syncthreads();
    const float nn = Red[2 * j] + Red[2 * j + 1];

    if (it == 2) {  // final pass: output norm of squashed vector
      if ((t & 1) == 0) {
        const float o = nn / (0.5f + nn);
        out[blockIdx.x * NCAP + j] = fminf(fmaxf(o, 1e-6f), 1.f - 1e-6f);
      }
      break;
    }

    const float scale = sqrtf(nn) / (0.5f + nn);  // squash factor
#pragma unroll
    for (int i = 0; i < 16; ++i) CC[j * DCAP + db + i] = s[i] * scale;
    __syncthreads();

    // logit update: b[n,j] += p[n,j,:] . cc[j,:]
    for (int idx = t; idx < NCAP * NCAP; idx += 256) {
      const float4* pr = (const float4*)(pb + (size_t)idx * DCAP);
      const float4* cv = (const float4*)(CC + (idx & (NCAP - 1)) * DCAP);
      float d = 0.f;
#pragma unroll
      for (int q = 0; q < 8; ++q) {
        const float4 a = pr[q], c = cv[q];
        d += a.x * c.x + a.y * c.y + a.z * c.z + a.w * c.w;
      }
      Blog[idx] += d;
    }
    __syncthreads();
  }
}

// ---------------------------------------------------------------------------
extern "C" void kernel_launch(void* const* d_in, const int* in_sizes, int n_in,
                              void* d_out, int out_size, void* d_ws,
                              size_t ws_size, hipStream_t stream) {
  (void)in_sizes; (void)n_in; (void)out_size; (void)ws_size;
  const float* x    = (const float*)d_in[0];  // [256,128,768]
  const float* W    = (const float*)d_in[1];  // [4096,768]
  const float* bias = (const float*)d_in[2];  // [1,128,4096]
  const float* l1w  = (const float*)d_in[3];  // [128,98304]
  const float* l1b  = (const float*)d_in[4];  // [128]
  const float* l2w  = (const float*)d_in[5];  // [128,128]
  const float* l2b  = (const float*)d_in[6];  // [128]

  float* out = (float*)d_out;                 // [256,128]
  float* p   = out + BATCH * NCAP;            // [256,128,128,32]
  float* r   = (float*)d_ws;                  // [256,128] relu hidden
  float* g   = r + BATCH * NCAP;              // [256,128] gate = 1+att

  attn1_kernel<<<dim3(NCAP / 16, BATCH / 16), 256, 0, stream>>>(x, l1w, l1b, r);
  attn2_kernel<<<BATCH, NCAP, 0, stream>>>(r, l2w, l2b, g);
  p_gemm_kernel<<<dim3(NOUT / 128, MROWS / 64), 256, 0, stream>>>(x, W, bias, g,
                                                                  p);
  const size_t smem = (size_t)(16384 + 16384 + 4096 + 256) * sizeof(float);
  routing_kernel<<<BATCH, 256, smem, stream>>>(p, out);
}
